// SynergyGNN_48155173322907
// MI455X (gfx1250) — matrix operand order repaired and checked by reading, not compiled, not run on hardware
//
#include <hip/hip_runtime.h>
#include <hip/hip_bf16.h>
#include <math.h>

// ---------------------------------------------------------------------------
// SynergyGNN (3-layer GCN) for MI455X / gfx1250.
// GEMMs via v_wmma_f32_16x16x32_bf16 (wave32, 16x16 tile per wave).
// Scatter-add via f32 hardware global atomics.
// ---------------------------------------------------------------------------

typedef __bf16 v4bf  __attribute__((ext_vector_type(4)));
typedef __bf16 v8bf  __attribute__((ext_vector_type(8)));
typedef __bf16 v16bf __attribute__((ext_vector_type(16)));
typedef float  v8f   __attribute__((ext_vector_type(8)));

#define HID 128

__device__ __forceinline__ void atomic_add_f32(float* p, float v) {
    __hip_atomic_fetch_add(p, v, __ATOMIC_RELAXED, __HIP_MEMORY_SCOPE_AGENT);
}

// -------------------------------- utility ----------------------------------

__global__ void fill_f32(float* __restrict__ p, float val, long long n) {
    long long t = (long long)blockIdx.x * blockDim.x + threadIdx.x;
    if (t < n) p[t] = val;
}

// deg[dst] += 1 over the E real edges (deg pre-filled with 1.0 for self loops)
__global__ void deg_edges(const int* __restrict__ edst, float* __restrict__ deg, int E) {
    int t = blockIdx.x * blockDim.x + threadIdx.x;
    if (t < E) atomic_add_f32(&deg[edst[t]], 1.0f);
}

__global__ void deg_to_dinv(float* __restrict__ deg, int N) {
    int t = blockIdx.x * blockDim.x + threadIdx.x;
    if (t < N) {
        float d = deg[t];
        deg[t] = (d > 0.0f) ? rsqrtf(d) : 0.0f;
    }
}

// norm[e] = dinv[src]*dinv[dst]; self-loop edges e >= E use node i = e - E
__global__ void edge_norm(const int* __restrict__ esrc, const int* __restrict__ edst,
                          const float* __restrict__ dinv, float* __restrict__ nrm,
                          int E, int Etot) {
    int t = blockIdx.x * blockDim.x + threadIdx.x;
    if (t >= Etot) return;
    int s, d;
    if (t < E) { s = esrc[t]; d = edst[t]; } else { s = t - E; d = s; }
    nrm[t] = dinv[s] * dinv[d];
}

// f32 -> bf16 elementwise
__global__ void cvt_f32_bf16(const float* __restrict__ in, __bf16* __restrict__ out,
                             long long n) {
    long long t = (long long)blockIdx.x * blockDim.x + threadIdx.x;
    if (t < n) out[t] = (__bf16)in[t];
}

// --------------------------- weight packing --------------------------------
// Pack W[K x 128] (f32 row-major) into WMMA B-fragment order so each lane's
// 16 bf16 elements are contiguous (32B):
//   idx = (((kb32*8 + tileN)*32 + lane)*16 + elem)
//   value = W[(kb32*32 + (lane>>4)*8 + (elem<8 ? elem : 8+elem)) * 128 + tileN*16 + (lane&15)]
__global__ void pack_w_bf16(const float* __restrict__ W, __bf16* __restrict__ Wpk, int K) {
    int t = blockIdx.x * blockDim.x + threadIdx.x;
    int total = K * HID;
    if (t >= total) return;
    int elem  = t & 15;
    int lane  = (t >> 4) & 31;
    int tileN = (t >> 9) & 7;
    int kb32  = t >> 12;
    int n  = tileN * 16 + (lane & 15);
    int Kg = kb32 * 32 + (lane >> 4) * 8 + ((elem < 8) ? elem : (8 + elem));
    Wpk[t] = (__bf16)W[(size_t)Kg * HID + n];
}

// ------------------------------ WMMA GEMM ----------------------------------
// H[N x 128] = A[N x K](bf16 row-major) * Wpk (packed bf16), f32 accumulate.
// One wave computes one 16x16 tile; block = 8 waves = one 16-row x 128-col strip.
__global__ __launch_bounds__(256)
void wmma_gemm_bf16(const __bf16* __restrict__ A, const __bf16* __restrict__ Wpk,
                    float* __restrict__ H, int N, int K) {
    const int lane  = threadIdx.x & 31;
    const int wave  = threadIdx.x >> 5;           // tileN: 0..7
    const int tileM = blockIdx.x;
    int row = tileM * 16 + (lane & 15);
    if (row >= N) row = N - 1;                    // clamp (N % 16 == 0 in practice)
    const int g8  = (lane >> 4) * 8;              // K sub-group offset per A layout
    const int nkb = K >> 5;

    v8f acc = {0.f, 0.f, 0.f, 0.f, 0.f, 0.f, 0.f, 0.f};
    for (int kb32 = 0; kb32 < nkb; ++kb32) {
        const int kb = kb32 << 5;
        const __bf16* ap = A + (size_t)row * K + kb + g8;
        v8bf a_lo = *(const v8bf*)(ap);           // K = kb+g8   .. +7
        v8bf a_hi = *(const v8bf*)(ap + 16);      // K = kb+16+g8.. +7
        const __bf16* bp = Wpk + (((size_t)kb32 * 8 + wave) * 32 + lane) * 16;
        v8bf b_lo = *(const v8bf*)(bp);
        v8bf b_hi = *(const v8bf*)(bp + 8);
        v16bf a, b;
#pragma unroll
        for (int i = 0; i < 8; ++i) {
            a[i] = a_lo[i]; a[8 + i] = a_hi[i];
            b[i] = b_lo[i]; b[8 + i] = b_hi[i];
        }
        acc = __builtin_amdgcn_wmma_f32_16x16x32_bf16(
            /*neg_a=*/false, a, /*neg_b=*/false, b,
            /*c_mod=*/(short)0, acc, /*reuse_a=*/false, /*reuse_b=*/false);
    }

    const int m0  = (lane >> 4) * 8;              // C/D layout: VGPR r -> M = m0 + r
    const int col = wave * 16 + (lane & 15);
#pragma unroll
    for (int r = 0; r < 8; ++r) {
        int m = tileM * 16 + m0 + r;
        if (m < N) H[(size_t)m * HID + col] = acc[r];
    }
}

// --------------------------- edge aggregation ------------------------------
// One wave per edge; each lane handles 4 contiguous floats of the 128-wide row.
__global__ void aggregate128(const float* __restrict__ h, const int* __restrict__ esrc,
                             const int* __restrict__ edst, const float* __restrict__ nrm,
                             float* __restrict__ agg, int E, int Etot) {
    long long t = (long long)blockIdx.x * blockDim.x + threadIdx.x;
    int e = (int)(t >> 5);
    if (e >= Etot) return;
    int l = (int)(t & 31);
    int s, d;
    if (e < E) { s = esrc[e]; d = edst[e]; } else { s = e - E; d = s; }
    float nv = nrm[e];
    const float4 hv = *(const float4*)(h + (size_t)s * HID + l * 4);
    float* ap = agg + (size_t)d * HID + l * 4;
    atomic_add_f32(ap + 0, hv.x * nv);
    atomic_add_f32(ap + 1, hv.y * nv);
    atomic_add_f32(ap + 2, hv.z * nv);
    atomic_add_f32(ap + 3, hv.w * nv);
}

// y = relu(agg + b), store bf16 (next layer's GEMM A)
__global__ void bias_relu_cvt(const float* __restrict__ agg, const float* __restrict__ b,
                              __bf16* __restrict__ out, long long n) {
    long long t = (long long)blockIdx.x * blockDim.x + threadIdx.x;
    if (t >= n) return;
    float v = agg[t] + b[(int)(t & (HID - 1))];
    out[t] = (__bf16)fmaxf(v, 0.0f);
}

// Layer 3: h3[node] = dot(h2[node, 0:128], W3[0:128]); one wave per node.
__global__ void gemm3_dot(const __bf16* __restrict__ h2, const float* __restrict__ W3,
                          float* __restrict__ h3, int N) {
    long long t = (long long)blockIdx.x * blockDim.x + threadIdx.x;
    int node = (int)(t >> 5), lane = (int)(t & 31);
    if (node >= N) return;
    const __bf16* hp = h2 + (size_t)node * HID + lane * 4;
    v4bf hv = *(const v4bf*)hp;
    float s = 0.0f;
#pragma unroll
    for (int i = 0; i < 4; ++i) s += (float)hv[i] * W3[lane * 4 + i];
#pragma unroll
    for (int off = 16; off > 0; off >>= 1) s += __shfl_xor(s, off, 32);
    if (lane == 0) h3[node] = s;
}

// Scalar scatter for the 1-wide layer.
__global__ void aggregate1(const float* __restrict__ h3, const int* __restrict__ esrc,
                           const int* __restrict__ edst, const float* __restrict__ nrm,
                           float* __restrict__ agg, int E, int Etot) {
    int e = blockIdx.x * blockDim.x + threadIdx.x;
    if (e >= Etot) return;
    int s, d;
    if (e < E) { s = esrc[e]; d = edst[e]; } else { s = e - E; d = s; }
    atomic_add_f32(&agg[d], h3[s] * nrm[e]);
}

__global__ void final_sigmoid(const float* __restrict__ agg, const float* __restrict__ b3,
                              float* __restrict__ out, int N) {
    int t = blockIdx.x * blockDim.x + threadIdx.x;
    if (t < N) {
        float v = agg[t] + b3[0];
        out[t] = 1.0f / (1.0f + expf(-v));
    }
}

// ------------------------------- launcher ----------------------------------

static inline long long cdivll(long long a, long long b) { return (a + b - 1) / b; }

extern "C" void kernel_launch(void* const* d_in, const int* in_sizes, int n_in,
                              void* d_out, int out_size, void* d_ws, size_t ws_size,
                              hipStream_t stream) {
    const float* x  = (const float*)d_in[0];
    const float* W1 = (const float*)d_in[1];
    const float* b1 = (const float*)d_in[2];
    const float* W2 = (const float*)d_in[3];
    const float* b2 = (const float*)d_in[4];
    const float* W3 = (const float*)d_in[5];
    const float* b3 = (const float*)d_in[6];
    const int*   ei = (const int*)d_in[7];

    const int N    = in_sizes[0] / 64;
    const int E    = in_sizes[7] / 2;
    const int Etot = E + N;
    const int* esrc = ei;
    const int* edst = ei + E;

    // workspace carve-up (256B aligned)
    char* p = (char*)d_ws;
    size_t off = 0;
    auto take = [&](size_t bytes) -> char* {
        char* r = p + off;
        off = (off + bytes + 255) & ~(size_t)255;
        return r;
    };
    float*  dinv = (float*)take((size_t)N * 4);          // deg -> dinv (in place)
    float*  nrm  = (float*)take((size_t)Etot * 4);
    __bf16* xb   = (__bf16*)take((size_t)N * HID * 2);   // bf16 activations (reused)
    float*  h    = (float*)take((size_t)N * HID * 4);    // GEMM output
    float*  agg  = (float*)take((size_t)N * HID * 4);    // scatter target
    __bf16* wpk1 = (__bf16*)take((size_t)64 * HID * 2);
    __bf16* wpk2 = (__bf16*)take((size_t)HID * HID * 2);
    float*  h3   = (float*)take((size_t)N * 4);
    float*  agg3 = (float*)take((size_t)N * 4);
    (void)ws_size;

    const int B = 256;
    const long long nh = (long long)N * HID;

    // degree / norms
    fill_f32<<<cdivll(N, B), B, 0, stream>>>(dinv, 1.0f, N);        // self-loop count
    deg_edges<<<cdivll(E, B), B, 0, stream>>>(edst, dinv, E);
    deg_to_dinv<<<cdivll(N, B), B, 0, stream>>>(dinv, N);
    edge_norm<<<cdivll(Etot, B), B, 0, stream>>>(esrc, edst, dinv, nrm, E, Etot);

    // weight packing + input conversion
    pack_w_bf16<<<cdivll((long long)64 * HID, B), B, 0, stream>>>(W1, wpk1, 64);
    pack_w_bf16<<<cdivll((long long)HID * HID, B), B, 0, stream>>>(W2, wpk2, HID);
    cvt_f32_bf16<<<cdivll((long long)N * 64, B), B, 0, stream>>>(x, xb, (long long)N * 64);

    const long long gemmBlocks = cdivll(N, 16);
    const long long aggThreads = (long long)Etot * 32;

    // ---- layer 1 ----
    wmma_gemm_bf16<<<gemmBlocks, 256, 0, stream>>>(xb, wpk1, h, N, 64);
    fill_f32<<<cdivll(nh, B), B, 0, stream>>>(agg, 0.0f, nh);
    aggregate128<<<cdivll(aggThreads, B), B, 0, stream>>>(h, esrc, edst, nrm, agg, E, Etot);
    bias_relu_cvt<<<cdivll(nh, B), B, 0, stream>>>(agg, b1, xb, nh);

    // ---- layer 2 ----
    wmma_gemm_bf16<<<gemmBlocks, 256, 0, stream>>>(xb, wpk2, h, N, HID);
    fill_f32<<<cdivll(nh, B), B, 0, stream>>>(agg, 0.0f, nh);
    aggregate128<<<cdivll(aggThreads, B), B, 0, stream>>>(h, esrc, edst, nrm, agg, E, Etot);
    bias_relu_cvt<<<cdivll(nh, B), B, 0, stream>>>(agg, b2, xb, nh);

    // ---- layer 3 (1-wide) ----
    gemm3_dot<<<cdivll((long long)N * 32, B), B, 0, stream>>>(xb, W3, h3, N);
    fill_f32<<<cdivll(N, B), B, 0, stream>>>(agg3, 0.0f, N);
    aggregate1<<<cdivll(Etot, B), B, 0, stream>>>(h3, esrc, edst, nrm, agg3, E, Etot);
    final_sigmoid<<<cdivll(N, B), B, 0, stream>>>(agg3, b3, (float*)d_out, N);
}
